// LogicGatedSNN_66211215835591
// MI455X (gfx1250) — matrix operand order repaired and checked by reading, not compiled, not run on hardware
//
#include <hip/hip_runtime.h>

typedef __attribute__((ext_vector_type(16))) _Float16 v16h;
typedef __attribute__((ext_vector_type(8)))  float    v8f;
typedef __attribute__((ext_vector_type(4)))  float    f32x4;

#define IN_F  8192
#define OUT_F 8192

// ---------------------------------------------------------------------------
// Kernel 1: binarized GEMV via WMMA + LIF update.
// Grid: OUT_F/16 blocks of 256 threads (8 waves). Block handles 16 output rows.
// Wave w covers K range [w*1024, (w+1)*1024) with 32 WMMA_F32_16X16X32_F16 ops.
// ---------------------------------------------------------------------------
__global__ __launch_bounds__(256) void snn_gemv_lif(
    const float* __restrict__ x,        // spike_input [IN_F]
    const float* __restrict__ syn,      // synapse_states [OUT_F, IN_F]
    const float* __restrict__ vmem_in,  // membrane_potential [OUT_F]
    const float* __restrict__ thr,      // adaptive_threshold [OUT_F]
    const float* __restrict__ refr,     // refractory_period [OUT_F]
    float* __restrict__ spikes_out,     // [OUT_F]
    float* __restrict__ vmem_out,       // [OUT_F]
    float* __restrict__ refr_out)       // [OUT_F]
{
    __shared__ _Float16 xh[IN_F];       // 16 KB: spike vector in f16
    __shared__ float    partial[16][8]; // [row-in-tile][wave]

    const int tid  = threadIdx.x;
    const int wave = tid >> 5;          // 0..7
    const int lane = tid & 31;
    const int hi   = lane >> 4;         // half-wave: 0 or 1
    const int m    = lane & 15;         // row within 16-row tile
    const int row_base = blockIdx.x * 16;

    // Stage x as f16 in LDS (exact: x is 0.0 or 1.0)
    for (int i = tid; i < IN_F; i += 256) xh[i] = (_Float16)x[i];
    __syncthreads();

    const int kBase = wave * (IN_F / 8);            // 1024 K per wave
    const float* rowp = syn + (size_t)(row_base + m) * IN_F;

    v8f acc = {};
    for (int ks = 0; ks < IN_F / 8; ks += 32) {
        const int k0 = kBase + ks;

        // ---- A fragment (16x32 f16): contiguous K layout (matches B)
        //  lanes 0-15 : row m, K = k0+{0..15}
        //  lanes 16-31: row m, K = k0+{16..31}
        const int ka = k0 + hi * 16;
        const f32x4* rp4 = (const f32x4*)(rowp + ka);
        f32x4 f0 = __builtin_nontemporal_load(rp4 + 0);  // K ka+0..3
        f32x4 f1 = __builtin_nontemporal_load(rp4 + 1);  // K ka+4..7
        f32x4 f2 = __builtin_nontemporal_load(rp4 + 2);  // K ka+8..11
        f32x4 f3 = __builtin_nontemporal_load(rp4 + 3);  // K ka+12..15
        v16h a;
        #pragma unroll
        for (int j = 0; j < 4; ++j) {
            a[j]      = (f0[j] > 50.0f) ? (_Float16)1.0f : (_Float16)0.0f;
            a[j + 4]  = (f1[j] > 50.0f) ? (_Float16)1.0f : (_Float16)0.0f;
            a[j + 8]  = (f2[j] > 50.0f) ? (_Float16)1.0f : (_Float16)0.0f;
            a[j + 12] = (f3[j] > 50.0f) ? (_Float16)1.0f : (_Float16)0.0f;
        }

        // ---- B fragment (32x16 f16), x replicated into every column:
        //  lanes 0-15 : K = k0+0..15, lanes 16-31: K = k0+16..31
        v16h b = *(const v16h*)(xh + k0 + hi * 16);

        acc = __builtin_amdgcn_wmma_f32_16x16x32_f16(
            /*neg_a=*/false, a, /*neg_b=*/false, b,
            /*c_mod=*/(short)0, acc, /*reuse_a=*/false, /*reuse_b=*/false);
    }

    // All D columns are identical. Lane 0 holds rows 0-7, lane 16 rows 8-15.
    if (lane == 0 || lane == 16) {
        #pragma unroll
        for (int r = 0; r < 8; ++r)
            partial[r + hi * 8][wave] = acc[r];
    }
    __syncthreads();

    // Wave 0, lanes 0-15: reduce over 8 K-partials, then LIF update per row.
    if (wave == 0 && lane < 16) {
        float dot = 0.0f;
        #pragma unroll
        for (int wv = 0; wv < 8; ++wv) dot += partial[lane][wv];

        const int row = row_base + lane;
        const float rp = refr[row];
        const float v  = vmem_in[row] * 0.5f + dot * (1.0f - rp * 0.5f);
        const float s  = (v >= thr[row]) ? 1.0f : 0.0f;
        spikes_out[row] = s;
        vmem_out[row]   = v * (1.0f - s);
        float nr = rp + s - 0.1f;
        refr_out[row] = fminf(fmaxf(nr, 0.0f), 1.0f);
    }
}

// ---------------------------------------------------------------------------
// Kernel 2: eligibility trace stream update (512 MB of the 512 MB traffic
// besides weights). One block per row; non-temporal b128 loads/stores so the
// 256 MB trace stream does not thrash the 192 MB L2 (x/spikes stay resident).
// ---------------------------------------------------------------------------
__global__ __launch_bounds__(256) void snn_trace(
    const float* __restrict__ x,         // [IN_F]
    const float* __restrict__ trace_in,  // [OUT_F, IN_F]
    const float* __restrict__ spikes,    // [OUT_F] (written by kernel 1)
    float* __restrict__ trace_out)       // [OUT_F, IN_F]
{
    const int r = blockIdx.x;
    const float s = spikes[r];
    const f32x4* tin  = (const f32x4*)(trace_in + (size_t)r * IN_F);
    f32x4*       tout = (f32x4*)(trace_out + (size_t)r * IN_F);
    const f32x4* x4   = (const f32x4*)x;

    for (int i = threadIdx.x; i < IN_F / 4; i += 256) {
        f32x4 t  = __builtin_nontemporal_load(tin + i);
        f32x4 xv = x4[i];
        #pragma unroll
        for (int j = 0; j < 4; ++j)
            t[j] = fminf(fmaxf(t[j] * 0.8f + s * xv[j], 0.0f), 5.0f);
        __builtin_nontemporal_store(t, tout + i);
    }
}

// ---------------------------------------------------------------------------
// Launch. Inputs (setup_inputs order):
//   0 spike_input [8192], 1 synapse_states [8192*8192], 2 membrane_potential,
//   3 adaptive_threshold, 4 eligibility_trace [8192*8192], 5 refractory_period
// Outputs flat-concatenated: spikes, new_membrane, new_trace, new_refractory.
// ---------------------------------------------------------------------------
extern "C" void kernel_launch(void* const* d_in, const int* in_sizes, int n_in,
                              void* d_out, int out_size, void* d_ws, size_t ws_size,
                              hipStream_t stream) {
    const float* x     = (const float*)d_in[0];
    const float* syn   = (const float*)d_in[1];
    const float* vmem  = (const float*)d_in[2];
    const float* thr   = (const float*)d_in[3];
    const float* trace = (const float*)d_in[4];
    const float* refr  = (const float*)d_in[5];

    float* out        = (float*)d_out;
    float* spikes_out = out;                                   // [OUT_F]
    float* vmem_out   = out + OUT_F;                           // [OUT_F]
    float* trace_out  = out + 2 * OUT_F;                       // [OUT_F*IN_F]
    float* refr_out   = out + 2 * OUT_F + (size_t)OUT_F * IN_F;// [OUT_F]

    snn_gemv_lif<<<OUT_F / 16, 256, 0, stream>>>(
        x, syn, vmem, thr, refr, spikes_out, vmem_out, refr_out);

    snn_trace<<<OUT_F, 256, 0, stream>>>(x, trace, spikes_out, trace_out);
}
